// CrossMultiHeadAttention_67156108640426
// MI455X (gfx1250) — matrix operand order repaired and checked by reading, not compiled, not run on hardware
//
#include <hip/hip_runtime.h>
#include <hip/hip_bf16.h>

// ---------------------------------------------------------------------------
// CDNA5 (gfx1250) bf16-WMMA cross-attention layer.
//   One-shot f32->bf16 conversion of activations & weights; all GEMMs
//   (QKV proj, attention QK^T / PV, output proj) on v_wmma_f32_16x16x32_bf16
//   with double-buffered async global->LDS staging (ASYNCcnt overlap).
// ---------------------------------------------------------------------------

#define EMBED   1024
#define HEADS   16
#define HEADDIM 64
#define SEQB    2      // batch
#define SEQL    2048   // LQ == LK

typedef __attribute__((ext_vector_type(16))) __bf16 v16bf;
typedef __attribute__((ext_vector_type(8)))  __bf16 v8bf;
typedef __attribute__((ext_vector_type(8)))  float  v8f;
typedef __attribute__((ext_vector_type(4)))  int    v4i;

__device__ __forceinline__ __bf16 to_bf16(float f) { return (__bf16)f; }

__device__ __forceinline__ v8f wmma_bf16(v16bf a, v16bf b, v8f c) {
    // (neg_a, A, neg_b, B, c_mod, C, reuse_a, reuse_b)
    return __builtin_amdgcn_wmma_f32_16x16x32_bf16(false, a, false, b,
                                                   (short)0, c, false, false);
}

// ---- async global -> LDS staging (CDNA5), guarded fallback ----------------
#if __has_builtin(__builtin_amdgcn_global_load_async_to_lds_b128)
#define ASYNC_LDS 1
__device__ __forceinline__ void stage16(const void* g, void* l) {
    // signature: (v4i AS1*, v4i AS3*, imm offset, imm cpol)
    __builtin_amdgcn_global_load_async_to_lds_b128(
        (__attribute__((address_space(1))) v4i*)g,
        (__attribute__((address_space(3))) v4i*)l, 0, 0);
}
__device__ __forceinline__ void stage_wait() {
    asm volatile("s_wait_asynccnt 0x0" ::: "memory");
}
#else
__device__ __forceinline__ void stage16(const void* g, void* l) {
    *(uint4*)l = *(const uint4*)g;
}
__device__ __forceinline__ void stage_wait() {}
#endif

// ---------------------------------------------------------------------------
// Bulk f32 -> bf16 conversion (O(N^2) pass, hoisted out of the GEMM hot loop)
// ---------------------------------------------------------------------------
__global__ __launch_bounds__(256)
void cvt_f32_bf16(const float* __restrict__ in, __bf16* __restrict__ out, int n) {
    const int i = (blockIdx.x * 256 + threadIdx.x) * 8;
    if (i >= n) return;
    float4 a = *(const float4*)(in + i);
    float4 b = *(const float4*)(in + i + 4);
    union { __bf16 h[8]; uint4 v; } u;
    u.h[0] = to_bf16(a.x); u.h[1] = to_bf16(a.y);
    u.h[2] = to_bf16(a.z); u.h[3] = to_bf16(a.w);
    u.h[4] = to_bf16(b.x); u.h[5] = to_bf16(b.y);
    u.h[6] = to_bf16(b.z); u.h[7] = to_bf16(b.w);
    *(uint4*)(out + i) = u.v;
}

// ---------------------------------------------------------------------------
// GEMM:  Out[M,N] = (X[M,K] * W[N,K]^T + bias[N]) * scale   (K = N = 1024)
//   X, W already bf16.  Block: 256 thr (8 waves), 128x128 tile, K-step 32,
//   double-buffered async LDS staging.  Wave = 32x64 (2x4 accums, 8 wmma).
//   OUT_MODE: 0 -> bf16 head-major [B,H,L,64]       (Q, K projections)
//             1 -> bf16 head-major transposed [B,H,64,L]  (V^T projection)
//             2 -> f32 row-major [M, EMBED]          (output projection)
// ---------------------------------------------------------------------------
template <int OUT_MODE>
__global__ __launch_bounds__(256)
void gemm_bf16_wmma(const __bf16* __restrict__ X, const __bf16* __restrict__ Wb,
                    const float* __restrict__ bias, void* __restrict__ Out,
                    float scale, int L) {
    __shared__ alignas(32) __bf16 As[2][128][32];
    __shared__ alignas(32) __bf16 Bs[2][128][32];

    const int tid  = threadIdx.x;
    const int m0   = blockIdx.x * 128;
    const int n0   = blockIdx.y * 128;
    const int wv   = tid >> 5;
    const int lane = tid & 31;
    const int half = lane >> 4;
    const int ln   = lane & 15;
    const int wm   = wv >> 1;   // 0..3  (M direction, 32 rows each)
    const int wn   = wv & 1;    // 0..1  (N direction, 64 cols each)

    const int srow = tid >> 1;          // 0..127
    const int scol = (tid & 1) * 16;    // 0 / 16

    const __bf16* xrow = X  + (size_t)(m0 + srow) * EMBED + scol;
    const __bf16* wrow = Wb + (size_t)(n0 + srow) * EMBED + scol;

    v8f acc[2][4];
#pragma unroll
    for (int i = 0; i < 2; ++i)
#pragma unroll
        for (int j = 0; j < 4; ++j) acc[i][j] = {};

    // prologue: stage K-tile 0 into buffer 0
    stage16(xrow,     &As[0][srow][scol]);
    stage16(xrow + 8, &As[0][srow][scol + 8]);
    stage16(wrow,     &Bs[0][srow][scol]);
    stage16(wrow + 8, &Bs[0][srow][scol + 8]);

    int cur = 0;
    for (int k0 = 0; k0 < EMBED; k0 += 32, cur ^= 1) {
        stage_wait();          // my async writes into buf[cur] have landed
        __syncthreads();       // everyone's writes landed; buf[cur^1] free
        if (k0 + 32 < EMBED) { // stage next tile while computing this one
            const __bf16* xn = xrow + k0 + 32;
            const __bf16* wn_ = wrow + k0 + 32;
            stage16(xn,      &As[cur ^ 1][srow][scol]);
            stage16(xn + 8,  &As[cur ^ 1][srow][scol + 8]);
            stage16(wn_,     &Bs[cur ^ 1][srow][scol]);
            stage16(wn_ + 8, &Bs[cur ^ 1][srow][scol + 8]);
        }

        // ---- fragments in contiguous K order (k = 16*half + s), same for A and B ----
        v16bf a[2], b[4];
#pragma unroll
        for (int mt = 0; mt < 2; ++mt) {
            const int r = wm * 32 + mt * 16 + ln;
            union { v16bf v; v8bf h[2]; } u;
            u.h[0] = *(const v8bf*)&As[cur][r][16 * half];     // K = 16*half..+7
            u.h[1] = *(const v8bf*)&As[cur][r][16 * half + 8]; // K = 16*half+8..+15
            a[mt] = u.v;
        }
#pragma unroll
        for (int nt = 0; nt < 4; ++nt) {
            const int r = wn * 64 + nt * 16 + ln;              // column index
            b[nt] = *(const v16bf*)&Bs[cur][r][16 * half];     // K = 16*half..+15
        }
#pragma unroll
        for (int mt = 0; mt < 2; ++mt)
#pragma unroll
            for (int nt = 0; nt < 4; ++nt)
                acc[mt][nt] = wmma_bf16(a[mt], b[nt], acc[mt][nt]);
    }

    // ---- epilogue: C/D layout row = r + 8*half, col = ln ----
#pragma unroll
    for (int mt = 0; mt < 2; ++mt) {
#pragma unroll
        for (int nt = 0; nt < 4; ++nt) {
            const int n   = n0 + wn * 64 + nt * 16 + ln;
            const float bv_ = bias[n];
#pragma unroll
            for (int r = 0; r < 8; ++r) {
                const int m = m0 + wm * 32 + mt * 16 + r + 8 * half;
                const float val = (acc[mt][nt][r] + bv_) * scale;
                if (OUT_MODE == 2) {
                    ((float*)Out)[(size_t)m * EMBED + n] = val;
                } else {
                    const int bidx = m / L, pos = m % L;
                    const int h = n >> 6, dd = n & 63;
                    size_t idx;
                    if (OUT_MODE == 0)
                        idx = ((size_t)(bidx * HEADS + h) * L + pos) * HEADDIM + dd;
                    else
                        idx = ((size_t)(bidx * HEADS + h) * HEADDIM + dd) * L + pos;
                    ((__bf16*)Out)[idx] = to_bf16(val);
                }
            }
        }
    }
}

// ---------------------------------------------------------------------------
// Flash-style attention:  ctx = softmax(Qs K^T) V  per (b,h); Qs pre-scaled
// by 1/sqrt(64) in the projection epilogue.  Block = 8 waves x 16 q-rows.
// Double-buffered K [32 keys][64 d] and V^T [64 d][32 keys] tiles in LDS.
// ---------------------------------------------------------------------------
__global__ __launch_bounds__(256)
void attn_flash_wmma(const __bf16* __restrict__ Q, const __bf16* __restrict__ Kh,
                     const __bf16* __restrict__ Vt, __bf16* __restrict__ ctx,
                     int LQ, int LK) {
    __shared__ alignas(32) __bf16 Ks[2][32][64];
    __shared__ alignas(32) __bf16 Vs[2][64][32];
    __shared__ alignas(32) __bf16 Ps[8][16][32];   // per-wave C->A relayout

    const int tid  = threadIdx.x;
    const int wv   = tid >> 5;
    const int lane = tid & 31;
    const int half = lane >> 4;
    const int ln   = lane & 15;
    const int bh   = blockIdx.y;           // b*HEADS + h
    const int q0   = blockIdx.x * 128;
    const int qrow = q0 + wv * 16 + ln;

    // Resident Q A-fragments (d = 0..31 and 32..63)
    v16bf qa[2];
    {
        const __bf16* qp = Q + ((size_t)bh * LQ + qrow) * HEADDIM;
#pragma unroll
        for (int d = 0; d < 2; ++d) {
            union { v16bf v; v8bf h[2]; } u;
            u.h[0] = *(const v8bf*)(qp + d * 32 + 16 * half);
            u.h[1] = *(const v8bf*)(qp + d * 32 + 16 * half + 8);
            qa[d] = u.v;
        }
    }

    v8f Oacc[4];
#pragma unroll
    for (int i = 0; i < 4; ++i) Oacc[i] = {};
    float mrow[8], lsum[8];
#pragma unroll
    for (int r = 0; r < 8; ++r) { mrow[r] = -3.0e38f; lsum[r] = 0.0f; }

    const int kkey = tid >> 3, kd = (tid & 7) * 8;   // K-tile fill mapping
    const int vd   = tid >> 2, vk = (tid & 3) * 8;   // V-tile fill mapping
    const __bf16* kbase = Kh + ((size_t)bh * LK + kkey) * HEADDIM + kd;
    const __bf16* vbase = Vt + ((size_t)bh * HEADDIM + vd) * LK + vk;

    // prologue: stage key-tile 0 into buffer 0
    stage16(kbase, &Ks[0][kkey][kd]);
    stage16(vbase, &Vs[0][vd][vk]);

    int cur = 0;
    for (int kt = 0; kt < LK; kt += 32, cur ^= 1) {
        stage_wait();          // my async writes into buf[cur] have landed
        __syncthreads();       // everyone's writes landed; buf[cur^1] free
        if (kt + 32 < LK) {    // stage next key tile during compute
            stage16(kbase + (size_t)(kt + 32) * HEADDIM, &Ks[cur ^ 1][kkey][kd]);
            stage16(vbase + (kt + 32),                   &Vs[cur ^ 1][vd][vk]);
        }

        // ---- S = Q K^T for 32 keys (two 16-key n-tiles) ----
        v8f s0 = {}, s1 = {};
#pragma unroll
        for (int d = 0; d < 2; ++d) {
            v16bf b0 = *(const v16bf*)&Ks[cur][ln][d * 32 + 16 * half];
            v16bf b1 = *(const v16bf*)&Ks[cur][16 + ln][d * 32 + 16 * half];
            s0 = wmma_bf16(qa[d], b0, s0);
            s1 = wmma_bf16(qa[d], b1, s1);
        }

        // ---- online softmax (each row lives within one 16-lane half) ----
        float p0[8], p1[8], t[8];
#pragma unroll
        for (int r = 0; r < 8; ++r) {
            p0[r] = s0[r];
            p1[r] = s1[r];
            t[r]  = fmaxf(p0[r], p1[r]);
        }
#pragma unroll
        for (int msk = 1; msk < 16; msk <<= 1)
#pragma unroll
            for (int r = 0; r < 8; ++r)
                t[r] = fmaxf(t[r], __shfl_xor(t[r], msk, 32));
        float alpha[8];
#pragma unroll
        for (int r = 0; r < 8; ++r) {
            const float mn = fmaxf(mrow[r], t[r]);
            alpha[r] = __expf(mrow[r] - mn);
            mrow[r]  = mn;
            p0[r] = __expf(p0[r] - mn);
            p1[r] = __expf(p1[r] - mn);
            t[r]  = p0[r] + p1[r];
        }
#pragma unroll
        for (int msk = 1; msk < 16; msk <<= 1)
#pragma unroll
            for (int r = 0; r < 8; ++r) t[r] += __shfl_xor(t[r], msk, 32);
#pragma unroll
        for (int r = 0; r < 8; ++r) lsum[r] = lsum[r] * alpha[r] + t[r];
#pragma unroll
        for (int nt = 0; nt < 4; ++nt)
#pragma unroll
            for (int r = 0; r < 8; ++r) Oacc[nt][r] *= alpha[r];

        // ---- C-layout -> A-layout via per-wave LDS (same-wave DS in order) --
#pragma unroll
        for (int r = 0; r < 8; ++r) {
            Ps[wv][r + 8 * half][ln]      = to_bf16(p0[r]);
            Ps[wv][r + 8 * half][16 + ln] = to_bf16(p1[r]);
        }
        union { v16bf v; v8bf h[2]; } up;
        up.h[0] = *(const v8bf*)&Ps[wv][ln][16 * half];
        up.h[1] = *(const v8bf*)&Ps[wv][ln][16 * half + 8];
        const v16bf pa = up.v;

        // ---- O += P V  (V^T tile: B column d is contiguous in keys) ----
#pragma unroll
        for (int nt = 0; nt < 4; ++nt) {
            v16bf vb = *(const v16bf*)&Vs[cur][nt * 16 + ln][16 * half];
            Oacc[nt] = wmma_bf16(pa, vb, Oacc[nt]);
        }
    }

    // ---- normalize and write ctx bf16 [B, LQ, EMBED] (heads re-merged) ----
    const int b_ = bh >> 4, h_ = bh & 15;
#pragma unroll
    for (int r = 0; r < 8; ++r) {
        const float inv = 1.0f / lsum[r];
        const int qq = q0 + wv * 16 + r + 8 * half;
#pragma unroll
        for (int nt = 0; nt < 4; ++nt) {
            const int col = h_ * HEADDIM + nt * 16 + ln;
            ctx[((size_t)b_ * LQ + qq) * EMBED + col] = to_bf16(Oacc[nt][r] * inv);
        }
    }
}

// ---------------------------------------------------------------------------
// out = LayerNorm(out + residual) * gamma + beta  (in-place on out, per row)
// ---------------------------------------------------------------------------
__global__ __launch_bounds__(256)
void resid_layernorm(float* __restrict__ out, const float* __restrict__ residual,
                     const float* __restrict__ gamma, const float* __restrict__ beta) {
    __shared__ float red[256];
    __shared__ float red2[256];
    const int row = blockIdx.x, tid = threadIdx.x;
    float x[4], s = 0.0f, s2 = 0.0f;
#pragma unroll
    for (int i = 0; i < 4; ++i) {
        const int c = tid + 256 * i;
        x[i] = out[(size_t)row * EMBED + c] + residual[(size_t)row * EMBED + c];
        s += x[i];
        s2 += x[i] * x[i];
    }
    red[tid] = s; red2[tid] = s2;
    __syncthreads();
    for (int off = 128; off > 0; off >>= 1) {
        if (tid < off) { red[tid] += red[tid + off]; red2[tid] += red2[tid + off]; }
        __syncthreads();
    }
    const float mean = red[0] * (1.0f / EMBED);
    const float var  = red2[0] * (1.0f / EMBED) - mean * mean;
    const float rstd = rsqrtf(var + 1e-5f);
#pragma unroll
    for (int i = 0; i < 4; ++i) {
        const int c = tid + 256 * i;
        out[(size_t)row * EMBED + c] = (x[i] - mean) * rstd * gamma[c] + beta[c];
    }
}

// ---------------------------------------------------------------------------
extern "C" void kernel_launch(void* const* d_in, const int* in_sizes, int n_in,
                              void* d_out, int out_size, void* d_ws, size_t ws_size,
                              hipStream_t stream) {
    const float* query = (const float*)d_in[0];   // [B, LQ, E] f32
    const float* kvseq = (const float*)d_in[1];   // [B, LK, E] f32
    // d_in[2] = prompt_size (unused by reference math)
    const float* Wq = (const float*)d_in[3];
    const float* bq = (const float*)d_in[4];
    const float* Wk = (const float*)d_in[5];
    const float* bk = (const float*)d_in[6];
    const float* Wv = (const float*)d_in[7];
    const float* bv = (const float*)d_in[8];
    const float* Wo = (const float*)d_in[9];
    const float* bo = (const float*)d_in[10];
    const float* ln_g = (const float*)d_in[11];
    const float* ln_b = (const float*)d_in[12];
    float* out = (float*)d_out;

    const int M = SEQB * SEQL;                                  // 4096 rows
    const size_t ACT = (size_t)M * EMBED;                       // 4.19M elems
    const size_t WEL = (size_t)EMBED * EMBED;                   // 1.05M elems

    __bf16* p = (__bf16*)d_ws;
    __bf16* Xq   = p; p += ACT;     // query bf16
    __bf16* Xkv  = p; p += ACT;     // kv bf16
    __bf16* Wqb  = p; p += WEL;
    __bf16* Wkb  = p; p += WEL;
    __bf16* Wvb  = p; p += WEL;
    __bf16* Wob  = p; p += WEL;
    __bf16* Qb   = p; p += ACT;     // [B,H,LQ,64], pre-scaled by 1/8
    __bf16* Kb   = p; p += ACT;     // [B,H,LK,64]
    __bf16* Vtb  = p; p += ACT;     // [B,H,64,LK]
    __bf16* ctxb = p; p += ACT;     // [B*LQ, E]
    (void)ws_size; (void)n_in; (void)in_sizes; (void)out_size;

    // ---- one-shot f32 -> bf16 conversions ----
    cvt_f32_bf16<<<(int)(ACT / 2048), 256, 0, stream>>>(query, Xq,  (int)ACT);
    cvt_f32_bf16<<<(int)(ACT / 2048), 256, 0, stream>>>(kvseq, Xkv, (int)ACT);
    cvt_f32_bf16<<<(int)(WEL / 2048), 256, 0, stream>>>(Wq, Wqb, (int)WEL);
    cvt_f32_bf16<<<(int)(WEL / 2048), 256, 0, stream>>>(Wk, Wkb, (int)WEL);
    cvt_f32_bf16<<<(int)(WEL / 2048), 256, 0, stream>>>(Wv, Wvb, (int)WEL);
    cvt_f32_bf16<<<(int)(WEL / 2048), 256, 0, stream>>>(Wo, Wob, (int)WEL);

    // ---- projections (Q pre-scaled by 1/sqrt(HEADDIM)) ----
    dim3 gg(M / 128, EMBED / 128);                 // (32, 8)
    gemm_bf16_wmma<0><<<gg, 256, 0, stream>>>(Xq,  Wqb, bq, Qb,  0.125f, SEQL);
    gemm_bf16_wmma<0><<<gg, 256, 0, stream>>>(Xkv, Wkb, bk, Kb,  1.0f,   SEQL);
    gemm_bf16_wmma<1><<<gg, 256, 0, stream>>>(Xkv, Wvb, bv, Vtb, 1.0f,   SEQL);

    // ---- attention ----
    dim3 ga(SEQL / 128, SEQB * HEADS);             // (16, 32)
    attn_flash_wmma<<<ga, 256, 0, stream>>>(Qb, Kb, Vtb, ctxb, SEQL, SEQL);

    // ---- output projection + residual LayerNorm ----
    gemm_bf16_wmma<2><<<gg, 256, 0, stream>>>(ctxb, Wob, bo, out, 1.0f, SEQL);
    resid_layernorm<<<M, 256, 0, stream>>>(out, query, ln_g, ln_b);
}
